// TensorParallelAttention_46712064312156
// MI455X (gfx1250) — matrix operand order repaired and checked by reading, not compiled, not run on hardware
//
#include <hip/hip_runtime.h>

typedef __attribute__((ext_vector_type(16))) __bf16 v16bf;
typedef __attribute__((ext_vector_type(8)))  float  v8f;
typedef __attribute__((ext_vector_type(8)))  unsigned short ushort8;

#define HIDDEN 2048
#define NHEADS 16
#define HDIM   128
#define SEQ    2048
#define BATCH  2

__device__ __forceinline__ unsigned short f2bf(float f) {
  unsigned int u = __float_as_uint(f);
  u += 0x7FFFu + ((u >> 16) & 1u);   // round-to-nearest-even
  return (unsigned short)(u >> 16);
}

union FragAB {
  v16bf   bf;
  ushort8 h[2];
};

// ---------------------------------------------------------------------------
// Tiled WMMA GEMM with bias:  C[M,N] = A[M,K] * B[K,N] + bias[N]
// A f32 or bf16 (template), B f32 (converted to bf16 while staging),
// C f32 or bf16 (template).  128x128 block tile, 8 waves of 32x64 each,
// BK=32, double-buffered LDS with register prefetch of the next K tile.
// ---------------------------------------------------------------------------
template<bool A_BF16, bool OUT_BF16>
__global__ __launch_bounds__(256)
void wmma_gemm_bias(const void* __restrict__ Ap, const float* __restrict__ B,
                    const float* __restrict__ bias, void* __restrict__ Cp,
                    int M, int N, int K)
{
  __shared__ __align__(16) unsigned short Asl[2][128][40];  // [buf][m][k]
  __shared__ __align__(16) unsigned short Bsl[2][128][40];  // [buf][n][k]

  const int tid  = threadIdx.x;
  const int lane = tid & 31;
  const int wid  = tid >> 5;
  const int l16  = lane & 15;
  const int hi   = (lane >> 4) & 1;

  const int nBase = blockIdx.x * 128;
  const int mBase = blockIdx.y * 128;
  const int waveM = (wid >> 1) * 32;
  const int waveN = (wid & 1) * 64;

  const v8f vzero = {};
  v8f acc[2][4];
  #pragma unroll
  for (int r = 0; r < 2; ++r)
    #pragma unroll
    for (int c = 0; c < 4; ++c) acc[r][c] = vzero;

  // register staging for next tile
  float4  aR[4];     // f32 A path: 4 x float4 per thread (128x32 tile)
  ushort8 aRh[2];    // bf16 A path: 2 x ushort8 per thread
  float4  bR[4];     // B: 4 x float4 per thread (32x128 tile)

  // ---- load tile kt into registers ----
  auto loadRegs = [&](int kt) {
    if (A_BF16) {
      const unsigned short* A = (const unsigned short*)Ap;
      #pragma unroll
      for (int j = 0; j < 2; ++j) {
        const int flat = j * 256 + tid;          // 512 x ushort8
        const int row  = flat >> 2;
        const int k    = (flat & 3) * 8;
        aRh[j] = *(const ushort8*)&A[(size_t)(mBase + row) * K + kt + k];
      }
    } else {
      const float* A = (const float*)Ap;
      #pragma unroll
      for (int j = 0; j < 4; ++j) {
        const int flat = j * 256 + tid;          // 1024 x float4
        const int row  = flat >> 3;
        const int k    = (flat & 7) * 4;
        aR[j] = *(const float4*)&A[(size_t)(mBase + row) * K + kt + k];
      }
    }
    #pragma unroll
    for (int j = 0; j < 4; ++j) {
      const int flat = j * 256 + tid;            // 1024 x float4
      const int k    = flat >> 5;
      const int n    = (flat & 31) * 4;
      bR[j] = *(const float4*)&B[(size_t)(kt + k) * N + nBase + n];
    }
  };

  // ---- store staged registers into LDS buffer ----
  auto storeRegs = [&](int buf) {
    if (A_BF16) {
      #pragma unroll
      for (int j = 0; j < 2; ++j) {
        const int flat = j * 256 + tid;
        const int row  = flat >> 2;
        const int k    = (flat & 3) * 8;
        *(ushort8*)&Asl[buf][row][k] = aRh[j];
      }
    } else {
      #pragma unroll
      for (int j = 0; j < 4; ++j) {
        const int flat = j * 256 + tid;
        const int row  = flat >> 3;
        const int k    = (flat & 7) * 4;
        Asl[buf][row][k + 0] = f2bf(aR[j].x);
        Asl[buf][row][k + 1] = f2bf(aR[j].y);
        Asl[buf][row][k + 2] = f2bf(aR[j].z);
        Asl[buf][row][k + 3] = f2bf(aR[j].w);
      }
    }
    #pragma unroll
    for (int j = 0; j < 4; ++j) {
      const int flat = j * 256 + tid;
      const int k    = flat >> 5;
      const int n    = (flat & 31) * 4;
      Bsl[buf][n + 0][k] = f2bf(bR[j].x);
      Bsl[buf][n + 1][k] = f2bf(bR[j].y);
      Bsl[buf][n + 2][k] = f2bf(bR[j].z);
      Bsl[buf][n + 3][k] = f2bf(bR[j].w);
    }
  };

  // prologue: stage first tile
  loadRegs(0);
  storeRegs(0);
  __syncthreads();

  int cur = 0;
  for (int kt = 0; kt < K; kt += 32) {
    const bool hasNext = (kt + 32) < K;
    if (hasNext) loadRegs(kt + 32);   // global loads overlap with WMMAs below

    FragAB a[2], b[4];
    #pragma unroll
    for (int r = 0; r < 2; ++r) {
      a[r].h[0] = *(const ushort8*)&Asl[cur][waveM + r * 16 + l16][hi * 8];
      a[r].h[1] = *(const ushort8*)&Asl[cur][waveM + r * 16 + l16][hi * 8 + 16];
    }
    #pragma unroll
    for (int c = 0; c < 4; ++c) {
      b[c].h[0] = *(const ushort8*)&Bsl[cur][waveN + c * 16 + l16][hi * 8];
      b[c].h[1] = *(const ushort8*)&Bsl[cur][waveN + c * 16 + l16][hi * 8 + 16];
    }
    #pragma unroll
    for (int r = 0; r < 2; ++r)
      #pragma unroll
      for (int c = 0; c < 4; ++c)
        acc[r][c] = __builtin_amdgcn_wmma_f32_16x16x32_bf16(
            false, a[r].bf, false, b[c].bf, (short)0, acc[r][c], false, false);

    if (hasNext) {
      storeRegs(cur ^ 1);
      __syncthreads();
    }
    cur ^= 1;
  }

  // ---- epilogue: bias add, store ----
  #pragma unroll
  for (int r = 0; r < 2; ++r) {
    #pragma unroll
    for (int c = 0; c < 4; ++c) {
      const int col = nBase + waveN + c * 16 + l16;
      const float bv = bias[col];
      #pragma unroll
      for (int i = 0; i < 8; ++i) {
        const int row = mBase + waveM + r * 16 + hi * 8 + i;
        const float v = acc[r][c][i] + bv;
        if (OUT_BF16) ((unsigned short*)Cp)[(size_t)row * N + col] = f2bf(v);
        else          ((float*)Cp)[(size_t)row * N + col] = v;
      }
    }
  }
}

// ---------------------------------------------------------------------------
// Flash attention over bf16 qkv [B*S, 3*HIDDEN].  One block per
// (128 q-rows, head, batch); 8 waves, each owning 16 q-rows.
// Online softmax in exp2 domain, WMMA for Q*K^T and P*V, register
// prefetch of the next K/V tile.  Output bf16 [B*S, HIDDEN].
// ---------------------------------------------------------------------------
__global__ __launch_bounds__(256)
void wmma_flash_attn(const unsigned short* __restrict__ qkv,
                     unsigned short* __restrict__ attn)
{
  __shared__ __align__(16) unsigned short Ksl[64][136];    // [key][d]
  __shared__ __align__(16) unsigned short Vtl[128][72];    // [d][key]  (V^T)
  __shared__ __align__(16) unsigned short Psl[8][16][72];  // per-wave P tile

  const int tid  = threadIdx.x;
  const int lane = tid & 31;
  const int wid  = tid >> 5;
  const int l16  = lane & 15;
  const int hi   = (lane >> 4) & 1;

  const int h     = blockIdx.y;
  const int b     = blockIdx.z;
  const int q0    = blockIdx.x * 128;
  const int rowg0 = b * SEQ;

  // softmax in base-2: fold 1/sqrt(128) * log2(e) into one scale
  const float scale2 = 0.088388347648318447f * 1.4426950408889634f;

  // ---- load Q fragments (wave-private 16 rows x 128 d) straight from global
  FragAB qf[4];
  {
    const int qrow = rowg0 + q0 + wid * 16 + l16;
    const unsigned short* qp = qkv + (size_t)qrow * (3 * HIDDEN) + h * HDIM;
    #pragma unroll
    for (int kk = 0; kk < 4; ++kk) {
      qf[kk].h[0] = *(const ushort8*)&qp[kk * 32 + hi * 8];
      qf[kk].h[1] = *(const ushort8*)&qp[kk * 32 + hi * 8 + 16];
    }
  }

  const v8f vzero = {};
  v8f o[8];
  #pragma unroll
  for (int t = 0; t < 8; ++t) o[t] = vzero;
  float mrow[8], lrow[8];
  #pragma unroll
  for (int i = 0; i < 8; ++i) { mrow[i] = -3.0e38f; lrow[i] = 0.f; }

  // K/V register staging: 4 x ushort8 each
  ushort8 kR[4], vR[4];
  const int key = tid >> 2;

  auto loadKV = [&](int kt) {
    const int krow = rowg0 + kt + key;
    const unsigned short* kp = qkv + (size_t)krow * (3 * HIDDEN) + HIDDEN + h * HDIM;
    const unsigned short* vp = qkv + (size_t)krow * (3 * HIDDEN) + 2 * HIDDEN + h * HDIM;
    #pragma unroll
    for (int j = 0; j < 4; ++j) {
      const int d = (tid & 3) * 32 + j * 8;
      kR[j] = *(const ushort8*)&kp[d];
      vR[j] = *(const ushort8*)&vp[d];
    }
  };
  auto storeKV = [&]() {
    #pragma unroll
    for (int j = 0; j < 4; ++j) {
      const int d = (tid & 3) * 32 + j * 8;
      *(ushort8*)&Ksl[key][d] = kR[j];
      #pragma unroll
      for (int e = 0; e < 8; ++e) Vtl[d + e][key] = vR[j][e];  // V transposed
    }
  };

  loadKV(0);
  storeKV();
  __syncthreads();

  for (int kt = 0; kt < SEQ; kt += 64) {
    const bool hasNext = (kt + 64) < SEQ;
    if (hasNext) loadKV(kt + 64);   // global loads overlap compute below

    // ---- S = Q * K^T  (16 x 64 per wave) ----
    v8f s[4];
    #pragma unroll
    for (int c = 0; c < 4; ++c) s[c] = vzero;
    #pragma unroll
    for (int kk = 0; kk < 4; ++kk) {
      #pragma unroll
      for (int c = 0; c < 4; ++c) {
        FragAB kb;
        kb.h[0] = *(const ushort8*)&Ksl[c * 16 + l16][kk * 32 + hi * 8];
        kb.h[1] = *(const ushort8*)&Ksl[c * 16 + l16][kk * 32 + hi * 8 + 16];
        s[c] = __builtin_amdgcn_wmma_f32_16x16x32_bf16(
            false, qf[kk].bf, false, kb.bf, (short)0, s[c], false, false);
      }
    }

    // ---- online softmax update (base-2 domain; row = hi*8 + i) ----
    #pragma unroll
    for (int i = 0; i < 8; ++i) {
      #pragma unroll
      for (int c = 0; c < 4; ++c) s[c][i] *= scale2;
      float mx = -3.0e38f;
      #pragma unroll
      for (int c = 0; c < 4; ++c) mx = fmaxf(mx, s[c][i]);
      mx = fmaxf(mx, __shfl_xor(mx, 1, 32));
      mx = fmaxf(mx, __shfl_xor(mx, 2, 32));
      mx = fmaxf(mx, __shfl_xor(mx, 4, 32));
      mx = fmaxf(mx, __shfl_xor(mx, 8, 32));
      const float mnew  = fmaxf(mrow[i], mx);
      const float alpha = __builtin_amdgcn_exp2f(mrow[i] - mnew);
      float rs = 0.f;
      #pragma unroll
      for (int c = 0; c < 4; ++c) {
        const float p = __builtin_amdgcn_exp2f(s[c][i] - mnew);
        s[c][i] = p;
        rs += p;
      }
      rs += __shfl_xor(rs, 1, 32);
      rs += __shfl_xor(rs, 2, 32);
      rs += __shfl_xor(rs, 4, 32);
      rs += __shfl_xor(rs, 8, 32);
      lrow[i] = lrow[i] * alpha + rs;
      mrow[i] = mnew;
      #pragma unroll
      for (int t = 0; t < 8; ++t) o[t][i] *= alpha;
    }

    // ---- reshape P (C-layout) -> A-layout via per-wave LDS tile ----
    #pragma unroll
    for (int c = 0; c < 4; ++c)
      #pragma unroll
      for (int i = 0; i < 8; ++i)
        Psl[wid][hi * 8 + i][c * 16 + l16] = f2bf(s[c][i]);
    asm volatile("s_wait_dscnt 0x0" ::: "memory");

    FragAB pa[2];
    #pragma unroll
    for (int kk2 = 0; kk2 < 2; ++kk2) {
      pa[kk2].h[0] = *(const ushort8*)&Psl[wid][l16][kk2 * 32 + hi * 8];
      pa[kk2].h[1] = *(const ushort8*)&Psl[wid][l16][kk2 * 32 + hi * 8 + 16];
    }
    // ---- O += P * V ----
    #pragma unroll
    for (int t = 0; t < 8; ++t) {
      #pragma unroll
      for (int kk2 = 0; kk2 < 2; ++kk2) {
        FragAB vb;
        vb.h[0] = *(const ushort8*)&Vtl[t * 16 + l16][kk2 * 32 + hi * 8];
        vb.h[1] = *(const ushort8*)&Vtl[t * 16 + l16][kk2 * 32 + hi * 8 + 16];
        o[t] = __builtin_amdgcn_wmma_f32_16x16x32_bf16(
            false, pa[kk2].bf, false, vb.bf, (short)0, o[t], false, false);
      }
    }

    if (hasNext) {
      __syncthreads();     // everyone done reading Ksl/Vtl
      storeKV();
      __syncthreads();     // new tile visible
    }
  }

  // ---- normalize and store (merged heads: [B*S, HIDDEN]) ----
  float linv[8];
  #pragma unroll
  for (int i = 0; i < 8; ++i) linv[i] = 1.0f / lrow[i];
  #pragma unroll
  for (int t = 0; t < 8; ++t) {
    #pragma unroll
    for (int i = 0; i < 8; ++i) {
      const int row = rowg0 + q0 + wid * 16 + hi * 8 + i;
      const int col = h * HDIM + t * 16 + l16;
      attn[(size_t)row * HIDDEN + col] = f2bf(o[t][i] * linv[i]);
    }
  }
}

// ---------------------------------------------------------------------------
extern "C" void kernel_launch(void* const* d_in, const int* in_sizes, int n_in,
                              void* d_out, int out_size, void* d_ws, size_t ws_size,
                              hipStream_t stream) {
  const float* x     = (const float*)d_in[0];  // [2, 2048, 2048]
  const float* w_qkv = (const float*)d_in[1];  // [2048, 6144]
  const float* b_qkv = (const float*)d_in[2];  // [6144]
  const float* w_out = (const float*)d_in[3];  // [2048, 2048]
  const float* b_out = (const float*)d_in[4];  // [2048]

  const int M = BATCH * SEQ;                   // 4096
  unsigned short* qkvb  = (unsigned short*)d_ws;            // [4096, 6144] bf16
  unsigned short* attnb = qkvb + (size_t)M * 3 * HIDDEN;    // [4096, 2048] bf16

  // 1) qkv = x @ w_qkv + b_qkv   (f32 in, bf16 out)
  wmma_gemm_bias<false, true>
      <<<dim3((3 * HIDDEN) / 128, M / 128), 256, 0, stream>>>(
          (const void*)x, w_qkv, b_qkv, (void*)qkvb, M, 3 * HIDDEN, HIDDEN);

  // 2) flash attention (bf16 in/out)
  wmma_flash_attn<<<dim3(SEQ / 128, NHEADS, BATCH), 256, 0, stream>>>(qkvb, attnb);

  // 3) out = attn @ w_out + b_out   (bf16 A, f32 out)
  wmma_gemm_bias<true, false>
      <<<dim3(HIDDEN / 128, M / 128), 256, 0, stream>>>(
          (const void*)attnb, w_out, b_out, d_out, M, HIDDEN, HIDDEN);
}
